// FocalLoss_34703335751859
// MI455X (gfx1250) — compile-verified, matrix operand order checked by reading, not run on hardware
//
#include <hip/hip_runtime.h>

// Focal loss (gamma = 2), N x 32 logits, mean reduction.
// Memory-bound: ~132 MB @ 23.3 TB/s => ~5.7 us floor. Design = streaming
// 128-bit loads (2x tile unroll, 8 independent B128 loads in flight per lane),
// WMMA f16 ones-matvec for the 16-wide row sum-of-exp, ds_swizzle cross-lane
// exchange, L2 prefetch two strides ahead, deterministic two-stage reduction.

typedef __attribute__((ext_vector_type(16))) _Float16 v16h;
typedef __attribute__((ext_vector_type(8)))  float    v8f;

// ds_swizzle group-of-32 encodings: offset = {xor[14:10], or[9:5], and[4:0]}
#define SWZ_X1  0x041F
#define SWZ_X2  0x081F
#define SWZ_X4  0x101F
#define SWZ_X8  0x201F
#define SWZ_X16 0x401F

template <int IMM>
__device__ __forceinline__ float swz(float x) {
    return __int_as_float(__builtin_amdgcn_ds_swizzle(__float_as_int(x), IMM));
}

// Per-tile body. Must be called from wave-uniform control flow (WMMA needs
// EXEC == all ones). Lanes L and L+16 co-own row (L&15) of the 16-row tile.
__device__ __forceinline__ float focal_tile(
    float4 a0, float4 a1, float4 a2, float4 a3,
    int row, int lane, int half,
    const float* __restrict__ inp,
    const int*   __restrict__ tgt,
    const float* __restrict__ alpha,
    const v16h&  Bones)
{
    constexpr int C = 32;

    // Row max: local 16-wide max, then exchange with partner lane (xor 16).
    float mx = fmaxf(fmaxf(fmaxf(a0.x, a0.y), fmaxf(a0.z, a0.w)),
               fmaxf(fmaxf(fmaxf(a1.x, a1.y), fmaxf(a1.z, a1.w)),
               fmaxf(fmaxf(fmaxf(a2.x, a2.y), fmaxf(a2.z, a2.w)),
                     fmaxf(fmaxf(a3.x, a3.y), fmaxf(a3.z, a3.w)))));
    float rowmax = fmaxf(mx, swz<SWZ_X16>(mx));

    // exp(x - rowmax) packed directly into the 16x32 f16 A-matrix layout:
    // lanes 0-15 hold K=0-7,16-23 of row m; lanes 16-31 hold K=8-15,24-31.
    v16h A;
    A[0]  = (_Float16)__expf(a0.x - rowmax);
    A[1]  = (_Float16)__expf(a0.y - rowmax);
    A[2]  = (_Float16)__expf(a0.z - rowmax);
    A[3]  = (_Float16)__expf(a0.w - rowmax);
    A[4]  = (_Float16)__expf(a1.x - rowmax);
    A[5]  = (_Float16)__expf(a1.y - rowmax);
    A[6]  = (_Float16)__expf(a1.z - rowmax);
    A[7]  = (_Float16)__expf(a1.w - rowmax);
    A[8]  = (_Float16)__expf(a2.x - rowmax);
    A[9]  = (_Float16)__expf(a2.y - rowmax);
    A[10] = (_Float16)__expf(a2.z - rowmax);
    A[11] = (_Float16)__expf(a2.w - rowmax);
    A[12] = (_Float16)__expf(a3.x - rowmax);
    A[13] = (_Float16)__expf(a3.y - rowmax);
    A[14] = (_Float16)__expf(a3.z - rowmax);
    A[15] = (_Float16)__expf(a3.w - rowmax);

    // D = A x ones + 0 : every column n of D holds sumexp(row m), f32 accum.
    v8f c = {};
    v8f d = __builtin_amdgcn_wmma_f32_16x16x32_f16(
                false, A, false, Bones, (short)0, c, false, false);

    // Extract sumexp for this lane's row from the C/D layout:
    // lane L<16: d[j] = D[j, L] (rows 0-7); lane L>=16: d[j] = D[8+j, L-16].
    const int k7 = lane & 7;
    float v = d[0];
    v = (k7 == 1) ? d[1] : v;
    v = (k7 == 2) ? d[2] : v;
    v = (k7 == 3) ? d[3] : v;
    v = (k7 == 4) ? d[4] : v;
    v = (k7 == 5) ? d[5] : v;
    v = (k7 == 6) ? d[6] : v;
    v = (k7 == 7) ? d[7] : v;
    float w = swz<SWZ_X16>(v);
    float sumexp = (((lane + 8) & 16) != 0) ? w : v;   // lanes 8..23 need partner

    // One lane per row finishes the scalar focal term.
    float f = 0.0f;
    if (half == 0) {
        int   tc = tgt[row];
        float xt = inp[(size_t)row * C + tc];          // L0 hit: line just loaded
        float ce = rowmax + __logf(sumexp) - xt;       // -log p_t
        float pt = __expf(-ce);
        float om = 1.0f - pt;
        f = alpha[tc] * om * om * ce;                  // gamma = 2
    }
    return f;
}

__global__ __launch_bounds__(256)
void focal_main(const float* __restrict__ inp,   // [N, 32]
                const int*   __restrict__ tgt,   // [N]
                const float* __restrict__ alpha, // [32]
                float*       __restrict__ partials,
                int ntiles)                      // N / 16
{
    constexpr int C = 32;
    const int lane = threadIdx.x & 31;
    const int wib  = threadIdx.x >> 5;
    const int m    = lane & 15;     // row within 16-row tile
    const int half = lane >> 4;     // 0: cols {0-7,16-23}; 1: cols {8-15,24-31}
    const int wavesPerBlk = blockDim.x >> 5;
    const int gwave  = blockIdx.x * wavesPerBlk + wib;
    const int stride = gridDim.x * wavesPerBlk;

    // B operand = all ones (16x16x32 f16): D[m,n] = sum_k A[m,k]
    v16h Bones;
#pragma unroll
    for (int i = 0; i < 16; ++i) Bones[i] = (_Float16)1.0f;

    const int o = half * 2;
    float acc = 0.0f;
    int t = gwave;

    // 2x-unrolled main loop: issue both tiles' 4x B128 loads before any
    // compute so each lane keeps 8 independent 128-bit loads in flight.
    for (; t + stride < ntiles; t += 2 * stride) {
        const int row0 = (t << 4) + m;
        const int row1 = ((t + stride) << 4) + m;
        const float4* p0 = (const float4*)(inp + (size_t)row0 * C);
        const float4* p1 = (const float4*)(inp + (size_t)row1 * C);

        float4 x0 = p0[o + 0];
        float4 x1 = p0[o + 1];
        float4 x2 = p0[o + 4];
        float4 x3 = p0[o + 5];
        float4 y0 = p1[o + 0];
        float4 y1 = p1[o + 1];
        float4 y2 = p1[o + 4];
        float4 y3 = p1[o + 5];

        // Prefetch two strides ahead into L2 (speculative; dropped if OOB-translated).
        if (t + 2 * stride < ntiles) {
            __builtin_prefetch((const char*)(inp + ((size_t)row0 + ((size_t)stride << 5)) * C) + (size_t)o * 16, 0, 0);
            __builtin_prefetch((const char*)(inp + ((size_t)row1 + ((size_t)stride << 5)) * C) + (size_t)o * 16, 0, 0);
        }

        acc += focal_tile(x0, x1, x2, x3, row0, lane, half, inp, tgt, alpha, Bones);
        acc += focal_tile(y0, y1, y2, y3, row1, lane, half, inp, tgt, alpha, Bones);
    }

    // Tail: at most one tile remaining per wave.
    for (; t < ntiles; t += stride) {
        const int row = (t << 4) + m;
        const float4* p = (const float4*)(inp + (size_t)row * C);
        float4 x0 = p[o + 0];
        float4 x1 = p[o + 1];
        float4 x2 = p[o + 4];
        float4 x3 = p[o + 5];
        acc += focal_tile(x0, x1, x2, x3, row, lane, half, inp, tgt, alpha, Bones);
    }

    // Deterministic wave reduction (xor tree over 32 lanes).
    acc += swz<SWZ_X1>(acc);
    acc += swz<SWZ_X2>(acc);
    acc += swz<SWZ_X4>(acc);
    acc += swz<SWZ_X8>(acc);
    acc += swz<SWZ_X16>(acc);

    __shared__ float smem[8];
    if (lane == 0) smem[wib] = acc;
    __syncthreads();
    if (threadIdx.x == 0) {
        float s = 0.0f;
        for (int i = 0; i < wavesPerBlk; ++i) s += smem[i];
        partials[blockIdx.x] = s;
    }
}

__global__ __launch_bounds__(256)
void focal_reduce(const float* __restrict__ partials, int n, float inv_n,
                  float* __restrict__ out)
{
    __shared__ float lds[256];
    float s = 0.0f;
    for (int i = threadIdx.x; i < n; i += 256) s += partials[i];
    lds[threadIdx.x] = s;
    __syncthreads();
    for (int o = 128; o > 0; o >>= 1) {
        if ((int)threadIdx.x < o) lds[threadIdx.x] += lds[threadIdx.x + o];
        __syncthreads();
    }
    if (threadIdx.x == 0) out[0] = lds[0] * inv_n;
}

extern "C" void kernel_launch(void* const* d_in, const int* in_sizes, int n_in,
                              void* d_out, int out_size, void* d_ws, size_t ws_size,
                              hipStream_t stream)
{
    const float* inp   = (const float*)d_in[0];
    const int*   tgt   = (const int*)  d_in[1];
    const float* alpha = (const float*)d_in[2];
    float* out      = (float*)d_out;
    float* partials = (float*)d_ws;

    const int n_rows = in_sizes[1];      // N (targets count); N % 16 == 0 here
    const int ntiles = n_rows >> 4;

    const int threads = 256;             // 8 wave32 waves / block
    const int blocks  = 1024;            // 8192 waves: ~32 MB of loads in flight

    focal_main<<<blocks, threads, 0, stream>>>(inp, tgt, alpha, partials, ntiles);
    focal_reduce<<<1, 256, 0, stream>>>(partials, blocks, 1.0f / (float)n_rows, out);
}
